// separate_model_7456063226231
// MI455X (gfx1250) — matrix operand order repaired and checked by reading, not compiled, not run on hardware
//
#include <hip/hip_runtime.h>
#include <hip/hip_bf16.h>

// ---------------------------------------------------------------------------
// Cross-attention block for MI455X (gfx1250, wave32, WMMA f32_16x16x32_f16).
// B=128, C=512, T=256 (H*W), S2=S+T=512, GROUPS=32.
// GEMMs: v_wmma_f32_16x16x32_f16, B-operand staged via async global->LDS
// copies (double-buffered, ASYNCcnt-tracked) and read back one f16 per K row.
// ---------------------------------------------------------------------------

typedef __attribute__((ext_vector_type(16))) _Float16 v16h;
typedef __attribute__((ext_vector_type(8)))  float    v8f;
typedef int v4i_b128 __attribute__((vector_size(16)));   // matches async-LDS builtin param

#define WMMA_F16(a, b, c) \
  __builtin_amdgcn_wmma_f32_16x16x32_f16(false, (a), false, (b), (short)0, (c), false, false)

// ------------------- async global->LDS staging (CDNA5) ---------------------
#if defined(__has_builtin)
#  if __has_builtin(__builtin_amdgcn_global_load_async_to_lds_b128)
#    define HAVE_ASYNC_LDS 1
#  endif
#endif
#ifndef HAVE_ASYNC_LDS
#  define HAVE_ASYNC_LDS 0
#endif

#if HAVE_ASYNC_LDS
#  if __has_builtin(__builtin_amdgcn_s_wait_asynccnt)
#    define WAIT_ASYNC(n) __builtin_amdgcn_s_wait_asynccnt(n)
#  else
#    define WAIT_ASYNC(n) asm volatile("s_wait_asynccnt %0" :: "i"(n) : "memory")
#  endif
#else
#  define WAIT_ASYNC(n) do { } while (0)
#endif

__device__ __forceinline__ void stage_b128(_Float16* lds_dst,
                                           const _Float16* gsrc) {
#if HAVE_ASYNC_LDS
  __builtin_amdgcn_global_load_async_to_lds_b128(
      (__attribute__((address_space(1))) v4i_b128*)gsrc,
      (__attribute__((address_space(3))) v4i_b128*)lds_dst,
      0, 0);
#else
  // synchronous fallback: 16-byte copy through VGPRs
  *(ulonglong2*)lds_dst = *(const ulonglong2*)gsrc;
#endif
}

namespace {
constexpr int BATCH = 128;
constexpr int C     = 512;
constexpr int T     = 256;   // H*W
constexpr int S2    = 512;   // S + T
constexpr float SCALE = 0.21022410381342863f;   // 512^(-1/4)
constexpr float EPSV  = 1e-5f;
}

// --------------------------- fragment loaders ------------------------------
// A-fragment (16x32 f16, ISA 7.12.2): lane = 16*half + r holds row M=r.
// VGPR j (pair a[2j],a[2j+1]) holds K = {half*8+2j, +1} for j<4,
// K = {16+half*8+2(j-4), +1} for j>=4.  A is row-major with leading dim lda.
__device__ __forceinline__ v16h load_a_rm(const _Float16* __restrict__ A,
                                          int lda, int m0, int k0) {
  const int lane = threadIdx.x & 31;
  const int half = lane >> 4, r = lane & 15;
  const _Float16* row = A + (size_t)(m0 + r) * lda + k0;
  v16h a;
#pragma unroll
  for (int j = 0; j < 8; ++j) {
    const int kb = (j < 4) ? (half * 8 + 2 * j) : (8 + half * 8 + 2 * j);
    a[2 * j]     = row[kb];
    a[2 * j + 1] = row[kb + 1];
  }
  return a;
}

// ---------------- K=512 GEMM core with LDS-staged B operand ----------------
// Block = 8 waves, each wave owns a 16x64 tile of a 128x64 block tile.
// Per K-step (32), the block stages the 32x64 B-slab into LDS (one async
// b128 per lane: 256 threads x 16B = 4KB), double-buffered.  Row stride is
// padded to 72 f16 (144B) so 32-lane fragment reads are 2-way-conflict
// optimal.  B-fragment: lane l holds column N=l%16, K in A's order (per-element reads).
// K-loop kept rolled: the x2 unroll only rotated accumulator registers
// (v_mov_b64 churn + hazard v_nops) without adding overlap beyond what the
// async double-buffer already provides.
__device__ __forceinline__ void gemm_k512_core(
    const _Float16* __restrict__ A, int lda, int m0,
    const _Float16* __restrict__ Bsrc, int ldb, int n0,
    v8f acc[4]) {
  __shared__ _Float16 bt[2][32][72];
  const int tid  = threadIdx.x;
  const int kr   = tid >> 3;           // 0..31 : K row within slab
  const int nc   = (tid & 7) * 8;      // 0..56 : 8-f16 (16B) column chunk
  const int lane = tid & 31;

  stage_b128(&bt[0][kr][nc], Bsrc + (size_t)kr * ldb + n0 + nc);

#pragma unroll 1
  for (int ks = 0; ks < 16; ++ks) {
    const int cur = ks & 1;
    if (ks < 15) {
      stage_b128(&bt[cur ^ 1][kr][nc],
                 Bsrc + (size_t)((ks + 1) * 32 + kr) * ldb + n0 + nc);
      WAIT_ASYNC(1);                  // oldest async (slab for this step) done
    } else {
      WAIT_ASYNC(0);
    }
    __syncthreads();                  // slab visible to all 8 waves

    const v16h a = load_a_rm(A, lda, m0, ks * 32);
#pragma unroll
    for (int t = 0; t < 4; ++t) {
      v16h bb;
      for (int s = 0; s < 16; ++s)
        bb[s] = bt[cur][(s < 8) ? (8 * (lane >> 4) + s) : (8 + 8 * (lane >> 4) + s)][16 * t + (lane & 15)];
      acc[t] = WMMA_F16(a, bb, acc[t]);
    }
    __syncthreads();                  // done reading before slab is reused
  }
}

// --------------------------- f32 -> f16 convert ----------------------------
__global__ void __launch_bounds__(256)
cvt_f32_to_f16_x4(const float* __restrict__ in, _Float16* __restrict__ out, int n4) {
  int i = blockIdx.x * blockDim.x + threadIdx.x;
  const int stride = gridDim.x * blockDim.x;
  for (; i < n4; i += stride) {
    const float4 v = ((const float4*)in)[i];
    out[4 * i + 0] = (_Float16)v.x;
    out[4 * i + 1] = (_Float16)v.y;
    out[4 * i + 2] = (_Float16)v.z;
    out[4 * i + 3] = (_Float16)v.w;
  }
}

// ------------------------------ GroupNorm ----------------------------------
// One block per (group, batch). Group = 16 channels x 256 t = 4096 floats.
__global__ void __launch_bounds__(256)
groupnorm_f16(const float* __restrict__ X, const float* __restrict__ gw,
              const float* __restrict__ gb, _Float16* __restrict__ H) {
  const int g = blockIdx.x, b = blockIdx.y;
  const int tid = threadIdx.x;
  const int c = g * 16 + (tid >> 4);
  const int t0 = (tid & 15) * 16;
  const float* p = X + ((size_t)b * C + c) * T + t0;

  float vals[16];
#pragma unroll
  for (int q = 0; q < 4; ++q) {
    const float4 v = ((const float4*)p)[q];
    vals[4 * q + 0] = v.x; vals[4 * q + 1] = v.y;
    vals[4 * q + 2] = v.z; vals[4 * q + 3] = v.w;
  }
  float s = 0.f, sq = 0.f;
#pragma unroll
  for (int q = 0; q < 16; ++q) { s += vals[q]; sq += vals[q] * vals[q]; }

  __shared__ float ssum[256], ssq[256];
  ssum[tid] = s; ssq[tid] = sq;
  __syncthreads();
  for (int off = 128; off > 0; off >>= 1) {
    if (tid < off) { ssum[tid] += ssum[tid + off]; ssq[tid] += ssq[tid + off]; }
    __syncthreads();
  }
  const float mean = ssum[0] * (1.0f / 4096.0f);
  const float var  = ssq[0] * (1.0f / 4096.0f) - mean * mean;
  const float inv  = rsqrtf(var + EPSV);
  const float sc   = gw[c] * inv;
  const float sh   = gb[c] - mean * sc;

  _Float16* o = H + ((size_t)b * C + c) * T + t0;
#pragma unroll
  for (int q = 0; q < 16; ++q) o[q] = (_Float16)(vals[q] * sc + sh);
}

// --------------------------- qkv GEMM + routing ----------------------------
//   m in [0,512)    -> Qt[b][t][m]        * SCALE   (transposed store)
//   m in [512,1024) -> K [b][m-512][256+t]* SCALE
//   m in [1024,1536)-> V [b][m-1024][256+t]
__global__ void __launch_bounds__(256)
gemm_qkv(const _Float16* __restrict__ W, const float* __restrict__ bias,
         const _Float16* __restrict__ H, _Float16* __restrict__ QT,
         _Float16* __restrict__ KM, _Float16* __restrict__ VM) {
  const int b = blockIdx.z;
  const int wave = threadIdx.x >> 5;
  const int m0 = blockIdx.x * 128 + wave * 16;
  const int n0 = blockIdx.y * 64;

  v8f acc[4] = {};
  gemm_k512_core(W, C, m0, H + (size_t)b * C * T, T, n0, acc);

  const int lane = threadIdx.x & 31;
  const int half = lane >> 4, r = lane & 15;
  const int region = m0 >> 9;       // 0:q 1:k 2:v
  const int mloc = m0 & 511;
#pragma unroll
  for (int t = 0; t < 4; ++t) {
    const int n = n0 + 16 * t + r;
#pragma unroll
    for (int j = 0; j < 8; ++j) {
      const int mj = j + 8 * half;
      const float v = acc[t][j] + bias[m0 + mj];
      if (region == 0)
        QT[((size_t)b * T + n) * C + (mloc + mj)] = (_Float16)(v * SCALE);
      else if (region == 1)
        KM[((size_t)b * C + (mloc + mj)) * S2 + T + n] = (_Float16)(v * SCALE);
      else
        VM[((size_t)b * C + (mloc + mj)) * S2 + T + n] = (_Float16)v;
    }
  }
}

// --------------------------- ekv GEMM + routing ----------------------------
// M=1024: m<512 -> K[b][m][t]*SCALE ; else -> V[b][m-512][t]
__global__ void __launch_bounds__(256)
gemm_ekv(const _Float16* __restrict__ W, const float* __restrict__ bias,
         const _Float16* __restrict__ Xc, _Float16* __restrict__ KM,
         _Float16* __restrict__ VM) {
  const int b = blockIdx.z;
  const int wave = threadIdx.x >> 5;
  const int m0 = blockIdx.x * 128 + wave * 16;
  const int n0 = blockIdx.y * 64;

  v8f acc[4] = {};
  gemm_k512_core(W, C, m0, Xc + (size_t)b * C * T, T, n0, acc);

  const int lane = threadIdx.x & 31;
  const int half = lane >> 4, r = lane & 15;
  const int region = m0 >> 9;       // 0:ek 1:ev
  const int mloc = m0 & 511;
#pragma unroll
  for (int t = 0; t < 4; ++t) {
    const int n = n0 + 16 * t + r;
#pragma unroll
    for (int j = 0; j < 8; ++j) {
      const int mj = j + 8 * half;
      const float v = acc[t][j] + bias[m0 + mj];
      if (region == 0)
        KM[((size_t)b * C + (mloc + mj)) * S2 + n] = (_Float16)(v * SCALE);
      else
        VM[((size_t)b * C + (mloc + mj)) * S2 + n] = (_Float16)v;
    }
  }
}

// ------------------------------ logits GEMM --------------------------------
// L[b][t][s] = sum_c Qt[b][t][c] * K[b][c][s]   (both pre-scaled)
__global__ void __launch_bounds__(256)
gemm_logits(const _Float16* __restrict__ QT, const _Float16* __restrict__ KM,
            float* __restrict__ L) {
  const int b = blockIdx.z;
  const int wave = threadIdx.x >> 5;
  const int m0 = blockIdx.x * 128 + wave * 16;   // t
  const int n0 = blockIdx.y * 64;                // s

  v8f acc[4] = {};
  gemm_k512_core(QT + (size_t)b * T * C, C, m0,
                 KM + (size_t)b * C * S2, S2, n0, acc);

  const int lane = threadIdx.x & 31;
  const int half = lane >> 4, r = lane & 15;
  float* Lb = L + (size_t)b * T * S2;
#pragma unroll
  for (int t = 0; t < 4; ++t)
#pragma unroll
    for (int j = 0; j < 8; ++j)
      Lb[(size_t)(m0 + j + 8 * half) * S2 + (n0 + 16 * t + r)] = acc[t][j];
}

// ------------------------------- softmax -----------------------------------
// Row (b,t): 512 logits -> probabilities, stored transposed Pt[b][s][t] f16.
__global__ void __launch_bounds__(256)
softmax_rows(const float* __restrict__ L, _Float16* __restrict__ PT) {
  const int row = blockIdx.x;       // b*T + t
  const int b = row >> 8, t = row & 255;
  const int tid = threadIdx.x;
  const float* p = L + (size_t)row * S2;
  const float e0 = p[tid], e1 = p[tid + 256];

  __shared__ float red[256];
  red[tid] = fmaxf(e0, e1);
  __syncthreads();
  for (int off = 128; off > 0; off >>= 1) {
    if (tid < off) red[tid] = fmaxf(red[tid], red[tid + off]);
    __syncthreads();
  }
  const float m = red[0];
  __syncthreads();

  const float x0 = __expf(e0 - m), x1 = __expf(e1 - m);
  red[tid] = x0 + x1;
  __syncthreads();
  for (int off = 128; off > 0; off >>= 1) {
    if (tid < off) red[tid] += red[tid + off];
    __syncthreads();
  }
  const float inv = 1.0f / red[0];

  PT[((size_t)b * S2 + tid)       * T + t] = (_Float16)(x0 * inv);
  PT[((size_t)b * S2 + tid + 256) * T + t] = (_Float16)(x1 * inv);
}

// ------------------------------ AV GEMM ------------------------------------
// a[b][c][t] = sum_s V[b][c][s] * Pt[b][s][t]
__global__ void __launch_bounds__(256)
gemm_attn(const _Float16* __restrict__ VM, const _Float16* __restrict__ PT,
          _Float16* __restrict__ AM) {
  const int b = blockIdx.z;
  const int wave = threadIdx.x >> 5;
  const int m0 = blockIdx.x * 128 + wave * 16;   // c
  const int n0 = blockIdx.y * 64;                // t

  v8f acc[4] = {};
  gemm_k512_core(VM + (size_t)b * C * S2, S2, m0,
                 PT + (size_t)b * S2 * T, T, n0, acc);

  const int lane = threadIdx.x & 31;
  const int half = lane >> 4, r = lane & 15;
  _Float16* Ab = AM + (size_t)b * C * T;
#pragma unroll
  for (int t = 0; t < 4; ++t)
#pragma unroll
    for (int j = 0; j < 8; ++j)
      Ab[(size_t)(m0 + j + 8 * half) * T + (n0 + 16 * t + r)] =
          (_Float16)acc[t][j];
}

// ---------------------- proj GEMM + residual epilogue ----------------------
// out[b][o][t] = 2*morph - x - (proj_w[o,:].a[b][:,t] + proj_b[o])
__global__ void __launch_bounds__(256)
gemm_proj_epilogue(const _Float16* __restrict__ W, const float* __restrict__ bias,
                   const _Float16* __restrict__ AM, const float* __restrict__ Xin,
                   const float* __restrict__ Morph, float* __restrict__ Out) {
  const int b = blockIdx.z;
  const int wave = threadIdx.x >> 5;
  const int m0 = blockIdx.x * 128 + wave * 16;   // o
  const int n0 = blockIdx.y * 64;                // t

  v8f acc[4] = {};
  gemm_k512_core(W, C, m0, AM + (size_t)b * C * T, T, n0, acc);

  const int lane = threadIdx.x & 31;
  const int half = lane >> 4, r = lane & 15;
#pragma unroll
  for (int t = 0; t < 4; ++t) {
    const int n = n0 + 16 * t + r;
#pragma unroll
    for (int j = 0; j < 8; ++j) {
      const int m = m0 + j + 8 * half;
      const size_t idx = ((size_t)b * C + m) * T + n;
      const float hout = acc[t][j] + bias[m];
      Out[idx] = 2.0f * Morph[idx] - Xin[idx] - hout;
    }
  }
}

// ------------------------------ launcher -----------------------------------
extern "C" void kernel_launch(void* const* d_in, const int* in_sizes, int n_in,
                              void* d_out, int out_size, void* d_ws, size_t ws_size,
                              hipStream_t stream) {
  (void)in_sizes; (void)n_in; (void)out_size; (void)ws_size;
  const float* criminal = (const float*)d_in[0];
  const float* morph    = (const float*)d_in[1];
  const float* norm_w   = (const float*)d_in[2];
  const float* norm_b   = (const float*)d_in[3];
  const float* qkv_w    = (const float*)d_in[4];
  const float* qkv_b    = (const float*)d_in[5];
  const float* ekv_w    = (const float*)d_in[6];
  const float* ekv_b    = (const float*)d_in[7];
  const float* proj_w   = (const float*)d_in[8];
  const float* proj_b   = (const float*)d_in[9];
  float* out = (float*)d_out;

  // Workspace layout (bytes). Later buffers alias dead earlier ones.
  char* ws = (char*)d_ws;
  size_t o = 0;
  _Float16* w16_qkv  = (_Float16*)(ws + o); o += (size_t)1536 * 512 * 2;   // 1.5 MB
  _Float16* w16_ekv  = (_Float16*)(ws + o); o += (size_t)1024 * 512 * 2;   // 1.0 MB
  _Float16* w16_proj = (_Float16*)(ws + o); o += (size_t)512  * 512 * 2;   // 0.5 MB
  _Float16* h16      = (_Float16*)(ws + o); o += (size_t)BATCH * C * T * 2;   // 32 MB
  _Float16* cond16   = (_Float16*)(ws + o); o += (size_t)BATCH * C * T * 2;   // 32 MB
  _Float16* qt       = (_Float16*)(ws + o); o += (size_t)BATCH * T * C * 2;   // 32 MB
  _Float16* km       = (_Float16*)(ws + o); o += (size_t)BATCH * C * S2 * 2;  // 64 MB
  _Float16* vm       = (_Float16*)(ws + o); o += (size_t)BATCH * C * S2 * 2;  // 64 MB
  // Aliases (exact containment, strictly ordered producer/consumer chain):
  float*    logits = (float*)h16;      // 64 MB over h16+cond16 (dead after ekv gemm)
  _Float16* pt     = qt;               // 32 MB (qt dead after logits gemm)
  _Float16* am     = km;               // 32 MB over km (dead after logits gemm)

  // 1) weight + cond conversion to f16
  cvt_f32_to_f16_x4<<<768,  256, 0, stream>>>(qkv_w,  w16_qkv,  1536 * 512 / 4);
  cvt_f32_to_f16_x4<<<512,  256, 0, stream>>>(ekv_w,  w16_ekv,  1024 * 512 / 4);
  cvt_f32_to_f16_x4<<<256,  256, 0, stream>>>(proj_w, w16_proj, 512 * 512 / 4);
  cvt_f32_to_f16_x4<<<4096, 256, 0, stream>>>(morph,  cond16,   BATCH * C * T / 4);

  // 2) GroupNorm -> h16
  groupnorm_f16<<<dim3(32, BATCH), 256, 0, stream>>>(criminal, norm_w, norm_b, h16);

  // 3) qkv / ekv GEMMs (write Qt scaled, K scaled, V)
  gemm_qkv<<<dim3(12, 4, BATCH), 256, 0, stream>>>(w16_qkv, qkv_b, h16, qt, km, vm);
  gemm_ekv<<<dim3(8,  4, BATCH), 256, 0, stream>>>(w16_ekv, ekv_b, cond16, km, vm);

  // 4) logits = Qt x K  (f32)
  gemm_logits<<<dim3(2, 8, BATCH), 256, 0, stream>>>(qt, km, logits);

  // 5) softmax -> Pt (transposed, f16)
  softmax_rows<<<BATCH * T, 256, 0, stream>>>(logits, pt);

  // 6) a = V x Pt
  gemm_attn<<<dim3(4, 4, BATCH), 256, 0, stream>>>(vm, pt, am);

  // 7) proj + residual epilogue -> out (f32)
  gemm_proj_epilogue<<<dim3(4, 4, BATCH), 256, 0, stream>>>(
      w16_proj, proj_b, am, criminal, morph, out);
}
